// PolyAttention_17351667876121
// MI455X (gfx1250) — matrix operand order repaired and checked by reading, not compiled, not run on hardware
//
#include <hip/hip_runtime.h>
#include <math.h>

// Problem dims (fixed by the reference)
#define B_   64
#define L_   2048
#define DIN  400
#define H_   200
#define C_   32
// Padded dims for 16x16x32 bf16 WMMA tiling
#define DP   416   // DIN padded to 13 K-chunks of 32
#define HP   224   // H padded to 7 K-chunks of 32 (14 N-tiles of 16)
#define LKC  64    // L / 32 K-chunks for the final GEMM
#define DNT  26    // DP / 16 N-tiles for the final GEMM

typedef __bf16 bf16_t;
typedef __attribute__((ext_vector_type(16))) bf16_t          v16bf;
typedef __attribute__((ext_vector_type(8)))  float           v8f;
typedef __attribute__((ext_vector_type(8)))  unsigned short  ushort8;

union Frag { v16bf v; ushort8 h[2]; };

__device__ __forceinline__ unsigned short f2bf(float f) {
  union { float f; unsigned u; } cv; cv.f = f;
  unsigned u = cv.u;
  u += 0x7FFFu + ((u >> 16) & 1u);   // round-to-nearest-even
  return (unsigned short)(u >> 16);
}

__device__ __forceinline__ v8f wmma_bf16(Frag a, Frag b, v8f c) {
  return __builtin_amdgcn_wmma_f32_16x16x32_bf16(
      /*neg_a=*/false, a.v, /*neg_b=*/false, b.v,
      /*c_mod=*/(short)0, c, /*reuse_a=*/false, /*reuse_b=*/false);
}

// ---------------------------------------------------------------------------
// Pack W_proj [DIN,H] -> B-fragment blocks, padded [DP x HP].
// Block (kc,nc): 32 lanes x 8 dwords; lane n, dword g holds
// (K = kc*32 + (n>=16?16:0) + 2g + {0,1}, N = nc*16 + n%16).
// ---------------------------------------------------------------------------
__global__ void __launch_bounds__(256) conv_w(const float* __restrict__ W,
                                              unsigned short* __restrict__ dst) {
  int idx = blockIdx.x * blockDim.x + threadIdx.x;
  if (idx >= (DP / 32) * (HP / 16) * 32 * 8) return;
  int g = idx & 7, lane = (idx >> 3) & 31, rest = idx >> 8;
  int nc = rest % (HP / 16), kc = rest / (HP / 16);
  int K = kc * 32 + (lane >> 4) * 16 + 2 * g;
  int N = nc * 16 + (lane & 15);
  float v0 = (K     < DIN && N < H_) ? W[(size_t)K       * H_ + N] : 0.f;
  float v1 = (K + 1 < DIN && N < H_) ? W[(size_t)(K + 1) * H_ + N] : 0.f;
  dst[(size_t)idx * 2]     = f2bf(v0);
  dst[(size_t)idx * 2 + 1] = f2bf(v1);
}

// context [C,H] used as B matrix with K=h, N=c  (i.e. ctx^T), padded K to HP.
__global__ void __launch_bounds__(256) conv_ctx(const float* __restrict__ ctx,
                                                unsigned short* __restrict__ dst) {
  int idx = blockIdx.x * blockDim.x + threadIdx.x;
  if (idx >= (HP / 32) * (C_ / 16) * 32 * 8) return;
  int g = idx & 7, lane = (idx >> 3) & 31, rest = idx >> 8;
  int nc = rest % (C_ / 16), kc = rest / (C_ / 16);
  int K = kc * 32 + (lane >> 4) * 16 + 2 * g;
  int N = nc * 16 + (lane & 15);
  float v0 = (K     < H_) ? ctx[(size_t)N * H_ + K]     : 0.f;
  float v1 = (K + 1 < H_) ? ctx[(size_t)N * H_ + K + 1] : 0.f;
  dst[(size_t)idx * 2]     = f2bf(v0);
  dst[(size_t)idx * 2 + 1] = f2bf(v1);
}

// ---------------------------------------------------------------------------
// Stage A: fused  scores[b,c,l] = tanh(x@W) @ ctx^T + lam*bias - (1-mask)*1e12
// One wave per 16-row L tile; 4 waves / block. Also emits the xswz B-fragment
// blocks for Stage C from the already-staged LDS x tiles (saves one full HBM
// pass over x: wave pair (2p, 2p+1) owns one 32-row l-chunk).
// ---------------------------------------------------------------------------
__global__ void __launch_bounds__(128) poly_scores(
    const float* __restrict__ x, const float* __restrict__ bias,
    const float* __restrict__ mask, const float* __restrict__ lamada,
    const unsigned short* __restrict__ wswz,
    const unsigned short* __restrict__ cswz,
    float* __restrict__ S,
    unsigned short* __restrict__ xswz) {
  __shared__ __align__(16) unsigned short xT[4][16][DP];   // 53248 B
  __shared__ __align__(16) unsigned short pT[4][16][32];   //  4096 B
  const int w = threadIdx.x >> 5, lane = threadIdx.x & 31;
  const int r = lane & 15, half = lane >> 4;
  const int tile = blockIdx.x * 4 + w;           // 0..8191
  const int b = tile >> 7, l0 = (tile & 127) << 4;

  // Stage 16x400 f32 tile -> 16xDP bf16 (zero-padded) in LDS
  const float* xrow = x + ((size_t)b * L_ + l0) * DIN;
  for (int i = lane; i < 16 * (DP / 4); i += 32) {
    int rr = i / (DP / 4), q = i - rr * (DP / 4), d = q * 4;
    float4 v = {0.f, 0.f, 0.f, 0.f};
    if (d < DIN) v = *(const float4*)(xrow + (size_t)rr * DIN + d);
    xT[w][rr][d + 0] = f2bf(v.x);
    xT[w][rr][d + 1] = f2bf(v.y);
    xT[w][rr][d + 2] = f2bf(v.z);
    xT[w][rr][d + 3] = f2bf(v.w);
  }
  __syncthreads();

  // Fused xswz emission: B-fragment blocks (K=l chunk of 32, N=d tile of 16).
  // Lane n's 16 contiguous ushorts are K_local = (n>=16?16:0)+0..15, which is
  // rows e of xT[pairBase + half]. Each wave covers 13 of the 26 d-tiles.
  {
    const int pairBase = w & ~1;
    const int sub = w & 1;
    const int kcg = ((blockIdx.x * 4 + pairBase) & 127) >> 1;  // l-chunk in batch
    const int ws_ = pairBase + half;
#pragma unroll 1
    for (int q = 0; q < DNT / 2; ++q) {
      const int nc = sub * (DNT / 2) + q;
      const int dcol = nc * 16 + r;
      ushort8 v0, v1;
#pragma unroll
      for (int e = 0; e < 8; ++e) v0[e] = xT[ws_][e][dcol];
#pragma unroll
      for (int e = 0; e < 8; ++e) v1[e] = xT[ws_][e + 8][dcol];
      unsigned short* dst =
          xswz + (((size_t)(b * LKC + kcg) * DNT + nc) * 32 + lane) * 16;
      *(ushort8*)dst       = v0;
      *(ushort8*)(dst + 8) = v1;
    }
  }

  v8f s0 = {}, s1 = {};                           // scores c=0..15 / 16..31
  for (int hc = 0; hc < HP / 32; ++hc) {
    v8f p0 = {}, p1 = {};                         // proj h-cols 0..15 / 16..31
    for (int kc = 0; kc < DP / 32; ++kc) {
      Frag a, b0, b1;
      const unsigned short* ap = &xT[w][r][kc * 32 + half * 16];
      a.h[0] = *(const ushort8*)ap;
      a.h[1] = *(const ushort8*)(ap + 8);
      const unsigned short* wp0 =
          wswz + (size_t)((kc * (HP / 16) + hc * 2 + 0) * 32 + lane) * 16;
      b0.h[0] = *(const ushort8*)wp0;
      b0.h[1] = *(const ushort8*)(wp0 + 8);
      const unsigned short* wp1 =
          wswz + (size_t)((kc * (HP / 16) + hc * 2 + 1) * 32 + lane) * 16;
      b1.h[0] = *(const ushort8*)wp1;
      b1.h[1] = *(const ushort8*)(wp1 + 8);
      p0 = wmma_bf16(a, b0, p0);
      p1 = wmma_bf16(a, b1, p1);
    }
    __syncthreads();
    // tanh + D-layout -> A-layout lane transpose through LDS
    for (int g = 0; g < 8; ++g) {
      int m = g + half * 8;
      pT[w][m][r]      = f2bf(tanhf(p0[g]));
      pT[w][m][r + 16] = f2bf(tanhf(p1[g]));
    }
    __syncthreads();
    Frag pa, c0, c1;
    const unsigned short* pp = &pT[w][r][half * 16];
    pa.h[0] = *(const ushort8*)pp;
    pa.h[1] = *(const ushort8*)(pp + 8);
    const unsigned short* cp0 = cswz + (size_t)((hc * 2 + 0) * 32 + lane) * 16;
    c0.h[0] = *(const ushort8*)cp0;
    c0.h[1] = *(const ushort8*)(cp0 + 8);
    const unsigned short* cp1 = cswz + (size_t)((hc * 2 + 1) * 32 + lane) * 16;
    c1.h[0] = *(const ushort8*)cp1;
    c1.h[1] = *(const ushort8*)(cp1 + 8);
    s0 = wmma_bf16(pa, c0, s0);
    s1 = wmma_bf16(pa, c1, s1);
  }

  const float lam = lamada[0];
  for (int g = 0; g < 8; ++g) {
    const int l = l0 + g + half * 8;
    const float mk = (1.f - mask[(size_t)b * L_ + l]) * 1e12f;
    {
      const int c = r;
      float sv = s0[g] + lam * bias[((size_t)b * L_ + l) * C_ + c] - mk;
      S[((size_t)b * C_ + c) * L_ + l] = sv;
    }
    {
      const int c = r + 16;
      float sv = s1[g] + lam * bias[((size_t)b * L_ + l) * C_ + c] - mk;
      S[((size_t)b * C_ + c) * L_ + l] = sv;
    }
  }
}

// ---------------------------------------------------------------------------
// Stage B: softmax over L per (b,c) row; emit bf16 weights row-major
// (row-major == A-fragment memory order for the final GEMM).
// Each thread owns 8 row elements in registers: S is read exactly once.
// ---------------------------------------------------------------------------
__global__ void __launch_bounds__(256) poly_softmax(const float* __restrict__ S,
                                                    unsigned short* __restrict__ wt) {
  __shared__ float red[256];
  const int row = blockIdx.x, tid = threadIdx.x;
  const float* s = S + (size_t)row * L_;
  float v[L_ / 256];
#pragma unroll
  for (int k = 0; k < L_ / 256; ++k) v[k] = s[tid + k * 256];
  float m = v[0];
#pragma unroll
  for (int k = 1; k < L_ / 256; ++k) m = fmaxf(m, v[k]);
  red[tid] = m; __syncthreads();
  for (int o = 128; o > 0; o >>= 1) {
    if (tid < o) red[tid] = fmaxf(red[tid], red[tid + o]);
    __syncthreads();
  }
  m = red[0]; __syncthreads();
  float sum = 0.f;
#pragma unroll
  for (int k = 0; k < L_ / 256; ++k) { v[k] = expf(v[k] - m); sum += v[k]; }
  red[tid] = sum; __syncthreads();
  for (int o = 128; o > 0; o >>= 1) {
    if (tid < o) red[tid] += red[tid + o];
    __syncthreads();
  }
  const float inv = 1.f / red[0];
  unsigned short* wrow = wt + (size_t)row * L_;
#pragma unroll
  for (int k = 0; k < L_ / 256; ++k) wrow[tid + k * 256] = f2bf(v[k] * inv);
}

// ---------------------------------------------------------------------------
// Stage C: out[b,c,d] = weights[b,c,:] @ x[b,:,d]   (K = L = 2048)
// One wave per (b, c-tile, d-tile); 4 waves / block.
// ---------------------------------------------------------------------------
__global__ void __launch_bounds__(128) poly_out(
    const unsigned short* __restrict__ wt,
    const unsigned short* __restrict__ xswz,
    float* __restrict__ out) {
  const int w = threadIdx.x >> 5, lane = threadIdx.x & 31;
  const int r = lane & 15, half = lane >> 4;
  const int wid = blockIdx.x * 4 + w;            // 0..3327
  const int nc = wid % DNT;
  const int t = wid / DNT;
  const int ct = t & 1, b = t >> 1;
  v8f acc = {};
  const unsigned short* arow = wt + (size_t)(b * C_ + ct * 16 + r) * L_;
  for (int kc = 0; kc < LKC; ++kc) {
    Frag a, bb;
    const unsigned short* ap = arow + kc * 32 + half * 16;
    a.h[0] = *(const ushort8*)ap;
    a.h[1] = *(const ushort8*)(ap + 8);
    const unsigned short* bp =
        xswz + (((size_t)(b * LKC + kc) * DNT + nc) * 32 + lane) * 16;
    bb.h[0] = *(const ushort8*)bp;
    bb.h[1] = *(const ushort8*)(bp + 8);
    if (kc + 1 < LKC) __builtin_prefetch(bp + (size_t)DNT * 512, 0, 0);
    acc = wmma_bf16(a, bb, acc);
  }
  const int d = nc * 16 + r;
  if (d < DIN) {
    for (int g = 0; g < 8; ++g) {
      const int c = ct * 16 + g + half * 8;
      out[((size_t)b * C_ + c) * DIN + d] = acc[g];
    }
  }
}

// ---------------------------------------------------------------------------
extern "C" void kernel_launch(void* const* d_in, const int* in_sizes, int n_in,
                              void* d_out, int out_size, void* d_ws, size_t ws_size,
                              hipStream_t stream) {
  const float* x    = (const float*)d_in[0];
  const float* bias = (const float*)d_in[1];
  const float* mask = (const float*)d_in[2];
  const float* W    = (const float*)d_in[3];
  const float* ctx  = (const float*)d_in[4];
  const float* lam  = (const float*)d_in[5];
  float* out = (float*)d_out;

  // Workspace layout (bytes)
  const size_t szW   = (size_t)(DP / 32) * (HP / 16) * 1024;          //   186,368
  const size_t szCtx = (size_t)(HP / 32) * (C_ / 16) * 1024;          //    14,336
  const size_t szX   = (size_t)B_ * LKC * DNT * 1024;                 // 109,051,904
  const size_t szS   = (size_t)B_ * C_ * L_ * 4;                      //  16,777,216
  const size_t szWt  = (size_t)B_ * C_ * L_ * 2;                      //   8,388,608
  const size_t offW = 0, offC = offW + szW, offX = offC + szCtx,
               offS = offX + szX, offWt = offS + szS;
  if (ws_size < offWt + szWt) return;

  char* ws = (char*)d_ws;
  unsigned short* wswz = (unsigned short*)(ws + offW);
  unsigned short* cswz = (unsigned short*)(ws + offC);
  unsigned short* xswz = (unsigned short*)(ws + offX);
  float*          S    = (float*)(ws + offS);
  unsigned short* wt   = (unsigned short*)(ws + offWt);

  const int nW   = (DP / 32) * (HP / 16) * 256;     // 46,592
  const int nCtx = (HP / 32) * (C_ / 16) * 256;     //  3,584

  conv_w  <<<(nW   + 255) / 256, 256, 0, stream>>>(W,   wswz);
  conv_ctx<<<(nCtx + 255) / 256, 256, 0, stream>>>(ctx, cswz);
  poly_scores<<<(B_ * (L_ / 16)) / 4, 128, 0, stream>>>(x, bias, mask, lam,
                                                        wswz, cswz, S, xswz);
  poly_softmax<<<B_ * C_, 256, 0, stream>>>(S, wt);
  poly_out<<<(B_ * 2 * DNT) / 4, 128, 0, stream>>>(wt, xswz, out);
}
